// MSSetEncoder_88347477278886
// MI455X (gfx1250) — compile-verified, matrix operand order checked by reading, not run on hardware
//
#include <hip/hip_runtime.h>
#include <hip/hip_bf16.h>
#include <math.h>
#include <stdint.h>

#define BB    8
#define LL    512
#define DM    512
#define NH    8
#define DH    64
#define DFFN  2048
#define NL    4
#define ML    (BB*LL)          // 4096 rows
#define FILMH 64

// ---------------------------------------------------------------- WMMA types
typedef __attribute__((ext_vector_type(16))) __bf16 v16bf;
typedef __attribute__((ext_vector_type(2)))  __bf16 v2bf;
typedef __attribute__((ext_vector_type(8)))  float  v8f;

union Frag { v16bf v; unsigned u[8]; };

// native bf16 converters -> backend selects v_cvt_pk_bf16_f32-class ops
__device__ __forceinline__ unsigned short bf16u(float f) {
  __bf16 h = (__bf16)f;
  return __builtin_bit_cast(unsigned short, h);
}
__device__ __forceinline__ unsigned pack_bf16(float a, float b) {
  v2bf t;
  t.x = (__bf16)a;
  t.y = (__bf16)b;
  return __builtin_bit_cast(unsigned, t);
}
__device__ __forceinline__ v8f wmma_bf16(const Frag& a, const Frag& b, v8f c) {
  return __builtin_amdgcn_wmma_f32_16x16x32_bf16(false, a.v, false, b.v,
                                                 (short)0, c, false, false);
}
// K-offset inside a 32-wide slice for element-pair p (p=0..7), per ISA layout:
// lanes 0-15 hold K {0..7,16..23}, lanes 16-31 hold K {8..15,24..31}
__device__ __forceinline__ int koff_of(int p, int hi) {
  return ((p & 4) << 2) + (hi << 3) + ((p & 3) << 1);
}

// -------------------------------------------- async global->LDS DMA (CDNA5)
__device__ __forceinline__ void async_copy_b128(unsigned lds_addr, const void* gptr) {
  asm volatile("global_load_async_to_lds_b128 %0, %1, off"
               :: "v"(lds_addr), "v"((unsigned long long)(uintptr_t)gptr)
               : "memory");
}
__device__ __forceinline__ void async_wait0() {
  asm volatile("s_wait_asynccnt 0x0" ::: "memory");
}

// ------------------------------------------------------------ generic GEMM
// C[M,N] = A[M,K] @ W[K,N] + bias (optional ReLU); fp32 in/out, bf16 WMMA core
// workgroup tile 128x64, K-stage 64, per-wave tile 32x32 (4 accumulators,
// 8 WMMAs per barrier pair). A tile DMA'd to LDS via async b128 copies.
__global__ __launch_bounds__(256) void gemm_bf16_kernel(
    const float* __restrict__ A, const float* __restrict__ W,
    const float* __restrict__ bias, float* __restrict__ C,
    int M, int N, int K, int relu)
{
  __shared__ float sAf[128][64];          // fp32 A tile (async DMA)
  __shared__ unsigned short sB[64][64];   // bf16 W tile, [n][k]
  int ntiles = N >> 6;
  int m0 = (blockIdx.x / ntiles) * 128;
  int n0 = (blockIdx.x % ntiles) * 64;
  int t = threadIdx.x, lane = t & 31, w = t >> 5;
  int mo = (w >> 1) * 32;                 // wave's 32-row slice of tile
  int no = (w & 1) * 32;                  // wave's 32-col slice of tile
  int hi = lane >> 4, lm = lane & 15;

  v8f acc00 = {}, acc01 = {}, acc10 = {}, acc11 = {};
  for (int k0 = 0; k0 < K; k0 += 64) {
    // A tile: 128x64 fp32 = 8192 elems; 8 async b128 chunks per thread
#pragma unroll
    for (int j = 0; j < 8; ++j) {
      int c2 = t + j * 256;
      int r = c2 >> 4, kc = (c2 & 15) << 2;
      async_copy_b128((unsigned)(uintptr_t)(void*)&sAf[r][kc],
                      A + (size_t)(m0 + r) * K + k0 + kc);
    }
    // W tile: 64x64, transposed into [n][k] bf16 (coalesced over n)
#pragma unroll
    for (int j = 0; j < 16; ++j) {
      int i = t + j * 256;
      int n = i & 63, k = i >> 6;
      sB[n][k] = bf16u(W[(size_t)(k0 + k) * N + n0 + n]);
    }
    if (k0 + 64 < K)  // prefetch next W tile, one cacheline per thread
      __builtin_prefetch(W + (size_t)(k0 + 64 + (t >> 6)) * N + n0 + (t & 63), 0, 1);
    async_wait0();
    __syncthreads();
#pragma unroll
    for (int ks = 0; ks < 64; ks += 32) {
      Frag a0, a1, b0, b1;
#pragma unroll
      for (int p = 0; p < 8; ++p) {
        int ko = ks + koff_of(p, hi);
        a0.u[p] = pack_bf16(sAf[mo + lm][ko],      sAf[mo + lm][ko + 1]);
        a1.u[p] = pack_bf16(sAf[mo + 16 + lm][ko], sAf[mo + 16 + lm][ko + 1]);
        b0.u[p] = *(const unsigned*)&sB[no + lm][ko];
        b1.u[p] = *(const unsigned*)&sB[no + 16 + lm][ko];
      }
      acc00 = wmma_bf16(a0, b0, acc00);
      acc01 = wmma_bf16(a0, b1, acc01);
      acc10 = wmma_bf16(a1, b0, acc10);
      acc11 = wmma_bf16(a1, b1, acc11);
    }
    __syncthreads();
  }
#pragma unroll
  for (int r = 0; r < 8; ++r) {
    int row0 = m0 + mo + r + (hi << 3);
    int row1 = row0 + 16;
    int c0 = n0 + no + lm, c1 = c0 + 16;
    float bv0 = bias[c0], bv1 = bias[c1];
    float v00 = acc00[r] + bv0, v01 = acc01[r] + bv1;
    float v10 = acc10[r] + bv0, v11 = acc11[r] + bv1;
    if (relu) {
      v00 = fmaxf(v00, 0.f); v01 = fmaxf(v01, 0.f);
      v10 = fmaxf(v10, 0.f); v11 = fmaxf(v11, 0.f);
    }
    C[(size_t)row0 * N + c0] = v00;
    C[(size_t)row0 * N + c1] = v01;
    C[(size_t)row1 * N + c0] = v10;
    C[(size_t)row1 * N + c1] = v11;
  }
}

// ---------------------------------------------------- fused attention (flash)
// one block per (b, head, 16-row q block); logits strip kept in LDS
__global__ __launch_bounds__(256) void attn_kernel(
    const float* __restrict__ Qf, const float* __restrict__ Kf,
    const float* __restrict__ Vf, const float* __restrict__ biasB,
    const float* __restrict__ mask, const float* __restrict__ allpad,
    float* __restrict__ out)
{
  __shared__ unsigned short sQ[16][64];
  __shared__ float sL[16][512];
  __shared__ float red[16][16];
  int blk = blockIdx.x;
  int qb = blk & 31; int h = (blk >> 5) & 7; int b = blk >> 8;
  int t = threadIdx.x, lane = t & 31, w = t >> 5;
  int hi = lane >> 4, lm = lane & 15;
  const float scale = 0.125f;             // 1/sqrt(64) folded into Q

  for (int i = t; i < 16 * 64; i += 256) {
    int m = i >> 6, d = i & 63;
    sQ[m][d] = bf16u(Qf[((size_t)b * LL + qb * 16 + m) * DM + h * DH + d] * scale);
  }
  __syncthreads();

  // logits: each wave handles 4 key tiles of 16
  for (int kb = w; kb < 32; kb += 8) {
    int key = kb * 16 + lm;
    const float* kp = Kf + ((size_t)b * LL + key) * DM + h * DH;
    Frag a0, a1, b0, b1;
#pragma unroll
    for (int p = 0; p < 8; ++p) {
      int ko = koff_of(p, hi);
      a0.u[p] = *(const unsigned*)&sQ[lm][ko];
      a1.u[p] = *(const unsigned*)&sQ[lm][32 + ko];
      b0.u[p] = pack_bf16(kp[ko], kp[ko + 1]);
      b1.u[p] = pack_bf16(kp[32 + ko], kp[32 + ko + 1]);
    }
    v8f acc = {};
    acc = wmma_bf16(a0, b0, acc);
    acc = wmma_bf16(a1, b1, acc);
#pragma unroll
    for (int r = 0; r < 8; ++r) {
      int qrow = r + (hi << 3);
      int q = qb * 16 + qrow;
      float v = acc[r] + biasB[((size_t)b * LL + q) * LL + key];
      if (mask[b * LL + key] != 0.f) v = -1e9f;
      sL[qrow][key] = v;
    }
  }
  __syncthreads();

  // softmax: 16 threads per row, two-level LDS reduction
  {
    int row = t >> 4, sub = t & 15;
    int c0 = sub * 32;
    float mx = -3.4e38f;
#pragma unroll
    for (int j = 0; j < 32; ++j) mx = fmaxf(mx, sL[row][c0 + j]);
    red[row][sub] = mx;
    __syncthreads();
    if (sub == 0) {
      float m2 = red[row][0];
#pragma unroll
      for (int j = 1; j < 16; ++j) m2 = fmaxf(m2, red[row][j]);
      red[row][0] = m2;
    }
    __syncthreads();
    mx = red[row][0];
    __syncthreads();
    float s = 0.f;
#pragma unroll
    for (int j = 0; j < 32; ++j) {
      float e = __expf(sL[row][c0 + j] - mx);
      sL[row][c0 + j] = e;
      s += e;
    }
    red[row][sub] = s;
    __syncthreads();
    if (sub == 0) {
      float s2 = 0.f;
#pragma unroll
      for (int j = 0; j < 16; ++j) s2 += red[row][j];
      red[row][0] = s2;
    }
    __syncthreads();
    float inv = (allpad[b] != 0.f) ? 0.f : 1.f / red[row][0];
#pragma unroll
    for (int j = 0; j < 32; ++j) sL[row][c0 + j] *= inv;
  }
  __syncthreads();

  // A @ V : waves 0..3 each own a 16-col d tile, 16 accumulate-chained WMMAs
  if (w < 4) {
    int nd = w * 16 + lm;
    const float* vbase = Vf + (size_t)b * LL * DM + h * DH + nd;
    v8f acc = {};
    for (int it = 0; it < 16; ++it) {
      int ks = it * 32;
      Frag a, bb;
#pragma unroll
      for (int p = 0; p < 8; ++p) {
        int ko = koff_of(p, hi);
        a.u[p]  = pack_bf16(sL[lm][ks + ko], sL[lm][ks + ko + 1]);
        bb.u[p] = pack_bf16(vbase[(size_t)(ks + ko) * DM],
                            vbase[(size_t)(ks + ko + 1) * DM]);
      }
      acc = wmma_bf16(a, bb, acc);
    }
#pragma unroll
    for (int r = 0; r < 8; ++r) {
      int q = qb * 16 + r + (hi << 3);
      out[((size_t)b * LL + q) * DM + h * DH + nd] = acc[r];
    }
  }
}

// ------------------------------------------------------------ small kernels
__global__ __launch_bounds__(256) void embed_prep_kernel(
    const float* __restrict__ ms, const float* __restrict__ centers,
    const float* __restrict__ logsig, float* __restrict__ phi,
    float* __restrict__ tlog, float* __restrict__ mask)
{
  int i = blockIdx.x * 256 + threadIdx.x;
  if (i >= ML) return;
  float mz = ms[i * 3 + 0], inten = ms[i * 3 + 1], z = ms[i * 3 + 2];
  for (int c = 0; c < 64; ++c) {
    float sg = fmaxf(__expf(logsig[c]), 1e-4f);
    float dd = mz - centers[c];
    phi[(size_t)i * 64 + c] = __expf(-dd * dd / (2.f * sg * sg));
  }
  tlog[i] = logf(1.f + fmaxf(inten, 0.f));
  mask[i] = (fabsf(mz) + fabsf(inten) + fabsf(z) == 0.f) ? 1.f : 0.f;
}

__global__ void allpad_kernel(const float* __restrict__ mask, float* __restrict__ allpad)
{
  int b = threadIdx.x;
  if (b >= BB) return;
  float all = 1.f;
  for (int l = 0; l < LL; ++l) if (mask[b * LL + l] == 0.f) all = 0.f;
  allpad[b] = all;
}

__global__ __launch_bounds__(256) void embed_post_kernel(
    const float* __restrict__ tlog, const float* __restrict__ gw,
    const float* __restrict__ gb, const float* __restrict__ allpad,
    float* __restrict__ x)
{
  int i = blockIdx.x * 256 + threadIdx.x;
  if (i >= ML * DM) return;
  int row = i / DM, d = i - row * DM, b = row / LL;
  float g = 1.f / (1.f + __expf(-(tlog[row] * gw[d] + gb[d])));
  float v = x[i] * (0.5f + g);
  if (allpad[b] != 0.f) v = 0.f;
  x[i] = v;
}

__global__ __launch_bounds__(256) void bias_kernel(
    const float* __restrict__ ms,
    const float* __restrict__ cD, const float* __restrict__ lsD,
    const float* __restrict__ wD, const float* __restrict__ scD,
    const float* __restrict__ cI, const float* __restrict__ lsI,
    const float* __restrict__ wI, const float* __restrict__ scI,
    float* __restrict__ bias)
{
  __shared__ float sc[64], si[64], sw[64], jc[8], ji[8], jw[8];
  int t = threadIdx.x;
  if (t < 64) {
    sc[t] = cD[t];
    float s = fmaxf(__expf(lsD[t]), 1e-4f);
    si[t] = -0.5f / (s * s);
    sw[t] = wD[t];
  }
  if (t < 8) {
    jc[t] = cI[t];
    float s = fmaxf(__expf(lsI[t]), 1e-4f);
    ji[t] = -0.5f / (s * s);
    jw[t] = wI[t];
  }
  __syncthreads();
  size_t i = (size_t)blockIdx.x * 256 + t;      // B*L*L elements
  int k = (int)(i % LL);
  size_t r = i / LL;
  int q = (int)(r % LL);
  int b = (int)(r / LL);
  float d = fabsf(ms[((size_t)b * LL + q) * 3] - ms[((size_t)b * LL + k) * 3]);
  float s1 = 0.f;
  for (int c = 0; c < 64; ++c) { float dd = d - sc[c]; s1 += __expf(dd * dd * si[c]) * sw[c]; }
  float s2 = 0.f;
  for (int c = 0; c < 8;  ++c) { float dd = d - jc[c]; s2 += __expf(dd * dd * ji[c]) * jw[c]; }
  bias[i] = scD[0] * s1 + scI[0] * s2;
}

__global__ void film_gb_kernel(
    const float* __restrict__ mw, const float* __restrict__ w1,
    const float* __restrict__ b1, const float* __restrict__ w2,
    const float* __restrict__ b2, float* __restrict__ gb)
{
  __shared__ float hid[FILMH];
  int b = blockIdx.x;
  float m = mw[b];
  if (threadIdx.x < FILMH)
    hid[threadIdx.x] = fmaxf(m * w1[threadIdx.x] + b1[threadIdx.x], 0.f);
  __syncthreads();
  for (int o = threadIdx.x; o < 2 * DM; o += 256) {
    float s = b2[o];
    for (int hh = 0; hh < FILMH; ++hh) s += hid[hh] * w2[hh * 2 * DM + o];
    gb[b * 2 * DM + o] = s;
  }
}

__global__ __launch_bounds__(256) void film_apply_kernel(
    const float* __restrict__ gb, const float* __restrict__ allpad,
    float* __restrict__ x)
{
  int i = blockIdx.x * 256 + threadIdx.x;
  if (i >= ML * DM) return;
  int b = i / (LL * DM);
  int d = i % DM;
  float gamma = gb[b * 2 * DM + d], beta = gb[b * 2 * DM + DM + d];
  if (allpad[b] != 0.f) { gamma = 0.f; beta = 0.f; }
  x[i] = x[i] * (1.f + gamma) + beta;
}

// residual + LayerNorm; one wave per row
__global__ __launch_bounds__(256) void ln_kernel(
    const float* __restrict__ x, const float* __restrict__ add,
    const float* __restrict__ g, const float* __restrict__ bta,
    const float* __restrict__ allpad, float* __restrict__ out, int zero_pad)
{
  int w = threadIdx.x >> 5, lane = threadIdx.x & 31;
  int row = blockIdx.x * 8 + w;
  int b = row / LL;
  const float* xp = x + (size_t)row * DM;
  const float* ap = add + (size_t)row * DM;
  float v[16];
  float s = 0.f;
#pragma unroll
  for (int i = 0; i < 16; ++i) { v[i] = xp[lane + i * 32] + ap[lane + i * 32]; s += v[i]; }
  for (int o = 16; o; o >>= 1) s += __shfl_xor(s, o, 32);
  float mean = s * (1.f / DM);
  float q = 0.f;
#pragma unroll
  for (int i = 0; i < 16; ++i) { float d = v[i] - mean; q += d * d; }
  for (int o = 16; o; o >>= 1) q += __shfl_xor(q, o, 32);
  float rstd = rsqrtf(q * (1.f / DM) + 1e-5f);
  float z = (zero_pad && allpad[b] != 0.f) ? 0.f : 1.f;
  float* op = out + (size_t)row * DM;
#pragma unroll
  for (int i = 0; i < 16; ++i) {
    int c = lane + i * 32;
    op[c] = z * (g[c] * (v[i] - mean) * rstd + bta[c]);
  }
}

__global__ __launch_bounds__(256) void output_kernel(
    const float* __restrict__ x, const float* __restrict__ mask,
    float* __restrict__ out, int out_size)
{
  int i = blockIdx.x * 256 + threadIdx.x;
  if (i < ML * DM && i < out_size) out[i] = x[i];
  if (i < ML && ML * DM + i < out_size) out[ML * DM + i] = mask[i];
}

// ---------------------------------------------------------------- host side
extern "C" void kernel_launch(void* const* d_in, const int* in_sizes, int n_in,
                              void* d_out, int out_size, void* d_ws, size_t ws_size,
                              hipStream_t stream) {
  (void)in_sizes; (void)n_in; (void)ws_size;
  // pytree leaf order (jax sorted-key flatten):
  // 0 mass_spec, 1 mw,
  // 2..5  bias_d {centers, log_sigma, out_w, scale}
  // 6..9  bias_iso {centers, log_sigma, out_w, scale}
  // 10+16*li .. : blocks[li] {ff1_b, ff1_w, ff2_b, ff2_w, k_b, k_w, ln1_b, ln1_g,
  //                           ln2_b, ln2_g, o_b, o_w, q_b, q_w, v_b, v_w}
  // 74+4*li ..  : films[li] {b1, b2, w1, w2}
  // 90 gate.b, 91 gate.w, 92 rbf centers, 93 rbf log_sigma, 94 proj_b, 95 proj_w
  const float* ms = (const float*)d_in[0];
  const float* mw = (const float*)d_in[1];

  float* ws = (float*)d_ws;
  size_t off = 0;
  float* phi    = ws + off; off += (size_t)ML * 64;
  float* tlog   = ws + off; off += ML;
  float* maskf  = ws + off; off += ML;
  float* allpad = ws + off; off += 64;
  float* gb     = ws + off; off += (size_t)BB * 2 * DM;
  float* x      = ws + off; off += (size_t)ML * DM;
  float* biasP  = ws + off; off += (size_t)BB * LL * LL;
  float* qb_    = ws + off; off += (size_t)ML * DM;
  float* kb_    = ws + off; off += (size_t)ML * DM;
  float* vb_    = ws + off; off += (size_t)ML * DM;
  float* attn   = ws + off; off += (size_t)ML * DM;
  float* tmp    = ws + off; off += (size_t)ML * DM;
  float* ffh    = ws + off; off += (size_t)ML * DFFN;

  // embedding
  embed_prep_kernel<<<ML / 256, 256, 0, stream>>>(ms, (const float*)d_in[92],
                                                  (const float*)d_in[93], phi, tlog, maskf);
  allpad_kernel<<<1, 32, 0, stream>>>(maskf, allpad);
  gemm_bf16_kernel<<<(ML / 128) * (DM / 64), 256, 0, stream>>>(
      phi, (const float*)d_in[95], (const float*)d_in[94], x, ML, DM, 64, 0);
  embed_post_kernel<<<(ML * DM) / 256, 256, 0, stream>>>(
      tlog, (const float*)d_in[91], (const float*)d_in[90], allpad, x);

  // pairwise bias
  bias_kernel<<<(BB * LL * LL) / 256, 256, 0, stream>>>(
      ms, (const float*)d_in[2], (const float*)d_in[3], (const float*)d_in[4],
      (const float*)d_in[5], (const float*)d_in[6], (const float*)d_in[7],
      (const float*)d_in[8], (const float*)d_in[9], biasP);

  for (int li = 0; li < NL; ++li) {
    int fb = 74 + 4 * li;
    int bb = 10 + 16 * li;
    film_gb_kernel<<<BB, 256, 0, stream>>>(
        mw, (const float*)d_in[fb + 2], (const float*)d_in[fb + 0],
        (const float*)d_in[fb + 3], (const float*)d_in[fb + 1], gb);
    film_apply_kernel<<<(ML * DM) / 256, 256, 0, stream>>>(gb, allpad, x);

    gemm_bf16_kernel<<<(ML / 128) * (DM / 64), 256, 0, stream>>>(
        x, (const float*)d_in[bb + 13], (const float*)d_in[bb + 12], qb_, ML, DM, DM, 0);
    gemm_bf16_kernel<<<(ML / 128) * (DM / 64), 256, 0, stream>>>(
        x, (const float*)d_in[bb + 5], (const float*)d_in[bb + 4], kb_, ML, DM, DM, 0);
    gemm_bf16_kernel<<<(ML / 128) * (DM / 64), 256, 0, stream>>>(
        x, (const float*)d_in[bb + 15], (const float*)d_in[bb + 14], vb_, ML, DM, DM, 0);

    attn_kernel<<<BB * NH * (LL / 16), 256, 0, stream>>>(
        qb_, kb_, vb_, biasP, maskf, allpad, attn);

    gemm_bf16_kernel<<<(ML / 128) * (DM / 64), 256, 0, stream>>>(
        attn, (const float*)d_in[bb + 11], (const float*)d_in[bb + 10], tmp, ML, DM, DM, 0);
    ln_kernel<<<ML / 8, 256, 0, stream>>>(
        x, tmp, (const float*)d_in[bb + 7], (const float*)d_in[bb + 6], allpad, x, 0);

    gemm_bf16_kernel<<<(ML / 128) * (DFFN / 64), 256, 0, stream>>>(
        x, (const float*)d_in[bb + 1], (const float*)d_in[bb + 0], ffh, ML, DFFN, DM, 1);
    gemm_bf16_kernel<<<(ML / 128) * (DM / 64), 256, 0, stream>>>(
        ffh, (const float*)d_in[bb + 3], (const float*)d_in[bb + 2], tmp, ML, DM, DFFN, 0);
    ln_kernel<<<ML / 8, 256, 0, stream>>>(
        x, tmp, (const float*)d_in[bb + 9], (const float*)d_in[bb + 8], allpad, x, 1);
  }

  output_kernel<<<(ML * DM) / 256, 256, 0, stream>>>(x, maskf, (float*)d_out, out_size);
}
